// SocialPooling_27513560498697
// MI455X (gfx1250) — matrix operand order repaired and checked by reading, not compiled, not run on hardware
//
#include <hip/hip_runtime.h>
#include <hip/hip_bf16.h>

// ---------------------------------------------------------------------------
// Social pooling + Linear + ReLU, fused, CDNA5 (gfx1250) wave32 WMMA bf16.
//
//   out[i,e] = relu( b[e] + sum_j C[j, bin(i,j), e] ) ,
//   C[j, b, e] = sum_h hs[j,h] * W[e, b*64+h]   (per-scene 64x64 @ 64x4096 GEMM)
//
// Round 3: ping-pong B double-buffer (no register-rotation copies) and
// row-pair packed C in LDS (v_cvt_pk_bf16_f32 + ds_store_b32, half the DS ops).
// ---------------------------------------------------------------------------

typedef __attribute__((ext_vector_type(16))) __bf16 v16bf;
typedef __attribute__((ext_vector_type(2)))  __bf16 v2bf;
typedef __attribute__((ext_vector_type(8)))  float  v8f;

#define N_SCENES 256
#define SCENE_LEN 64
#define H_DIM 64
#define EMB 64
#define COLS 4096              // (bin, emb) flattened: col = b*64 + e
#define NTILES 256             // COLS / 16
#define NFRAGS 512             // NTILES * 2 k-steps
#define FRAG_ELEMS 512         // 32 lanes * 16 bf16

__device__ __forceinline__ float bflo(unsigned int u) {   // low bf16 of dword
    u <<= 16;
    float f; __builtin_memcpy(&f, &u, 4); return f;
}
__device__ __forceinline__ float bfhi(unsigned int u) {   // high bf16 of dword
    u &= 0xffff0000u;
    float f; __builtin_memcpy(&f, &u, 4); return f;
}
__device__ __forceinline__ void acc4_lo(float* a, uint4 d) {
    a[0] += bflo(d.x); a[1] += bflo(d.y); a[2] += bflo(d.z); a[3] += bflo(d.w);
}
__device__ __forceinline__ void acc4_hi(float* a, uint4 d) {
    a[0] += bfhi(d.x); a[1] += bfhi(d.y); a[2] += bfhi(d.z); a[3] += bfhi(d.w);
}

// ---------------------------------------------------------------------------
// Prep: repack W[e, b*64+h] (f32, [64 x 4096]) into bf16 WMMA B-fragments,
// lane-contiguous. Fragment f = nt*2 + ks covers B tile (K = ks*32..+31) x
// (n = nt*16..+15) of Wr[h, b*64+e] = W[e, b*64+h]. k order used (contiguous, same as A): lanes
// 0-15 hold K=0..15 (2 per VGPR), lanes 16-31 hold K=16..31; n = lane & 15.
// ---------------------------------------------------------------------------
__global__ void __launch_bounds__(256) prep_w_frags(const float* __restrict__ W,
                                                    __bf16* __restrict__ wfrag) {
    int idx = blockIdx.x * 256 + threadIdx.x;         // 0 .. 512*512-1
    if (idx >= NFRAGS * FRAG_ELEMS) return;
    int f    = idx >> 9;          // fragment id
    int r    = idx & 511;         // element within fragment
    int lane = r >> 4;            // 0..31
    int q    = r & 15;            // 0..15 (VGPR-pair p, sub s)
    int nt   = f >> 1;
    int ks   = f & 1;
    int p    = q >> 1;
    int s    = q & 1;
    int kb   = (lane < 16) ? (2 * p) : (16 + 2 * p);
    int h    = ks * 32 + kb + s;                       // K index == hidden dim
    int col  = nt * 16 + (lane & 15);                  // (b,e) column
    int b    = col >> 6;
    int e    = col & 63;
    wfrag[idx] = (__bf16)W[e * COLS + b * 64 + h];
}

// ---------------------------------------------------------------------------
// Main: one workgroup (256 threads = 8 wave32) per scene.
// ---------------------------------------------------------------------------
__global__ void __launch_bounds__(256) social_pool_wmma(
    const float* __restrict__ hs_g,    // [N_PEDS, 64]
    const float* __restrict__ pos_g,   // [N_PEDS, 2]
    const __bf16* __restrict__ wfrag,  // [NFRAGS * FRAG_ELEMS]
    const float* __restrict__ bias,    // [64]
    float* __restrict__ out)           // [N_PEDS, 64]
{
    // C chunk, row-pair packed: Cs32[p*COLS + col] = {row 2p (lo), row 2p+1 (hi)}
    __shared__ __align__(16) unsigned int Cs32[8 * COLS];   // 128 KB
    __shared__ __align__(16) __bf16 hss[SCENE_LEN * H_DIM]; // 8 KB scene HS bf16
    __shared__ unsigned char binsS[SCENE_LEN * SCENE_LEN];  // 4 KB bin(i,j) | 255
    __shared__ float posS[SCENE_LEN * 2];                   // 512 B

    const int tid   = threadIdx.x;
    const int scene = blockIdx.x;
    const int lane  = tid & 31;
    const int wave  = tid >> 5;

    // ---- load positions + hidden states (convert to bf16) ----
    if (tid < 128) posS[tid] = pos_g[scene * 128 + tid];
    for (int k = tid; k < SCENE_LEN * H_DIM; k += 256)
        hss[k] = (__bf16)hs_g[scene * SCENE_LEN * H_DIM + k];
    __syncthreads();

    // ---- bin table: 16 (i,j) pairs per thread ----
    const float inv_norm = 7.0f / 4.0f;   // (GRID-1)/NEIGH
    #pragma unroll
    for (int r = 0; r < 16; ++r) {
        int pair = tid * 16 + r;          // 0..4095
        int i = pair >> 6, j = pair & 63;
        float rx = posS[2 * j]     - posS[2 * i];
        float ry = posS[2 * j + 1] - posS[2 * i + 1];
        float gxf = fminf(fmaxf(rx * inv_norm, -4.0f), 4.0f) + 4.0f;
        float gyf = fminf(fmaxf(ry * inv_norm, -4.0f), 4.0f) + 4.0f;
        int gx = (int)gxf, gy = (int)gyf; // trunc of non-negative == floor
        bool valid = (gx < 8) && (gy < 8) && (i != j);
        binsS[pair] = valid ? (unsigned char)(gy * 8 + gx) : (unsigned char)255;
    }
    __syncthreads();

    float acc[16];
    #pragma unroll
    for (int e = 0; e < 16; ++e) acc[e] = 0.0f;
    const int myi   = tid >> 2;           // ped index in scene (0..63)
    const int ebase = (tid & 3) << 4;     // this thread's 16 emb columns

    // Per-wave bases: wave handles n-tiles [wave*32, wave*32+32).
    const __bf16* __restrict__ fb = wfrag + (wave * 32) * 2 * FRAG_ELEMS + lane * 16;
    // store base: rowpair base (0 or 4) * COLS + column
    const int cbase32 = ((lane >= 16) ? 4 : 0) * COLS + (wave * 32) * 16 + (lane & 15);

    // ---- 4 chunks of 16 j-rows ----
    #pragma unroll 1
    for (int jc = 0; jc < 4; ++jc) {
        // Build A fragments (16x32 bf16, 2 k-steps) from LDS, contiguous k order (same as B):
        // lanes 0-15: V0-7=K0..15 ; lanes 16-31: V0-7=K16..31
        v16bf afr[2];
        {
            unsigned int* au = (unsigned int*)afr;
            int m = lane & 15;
            int j = jc * 16 + m;
            #pragma unroll
            for (int ks = 0; ks < 2; ++ks) {
                #pragma unroll
                for (int p = 0; p < 8; ++p) {
                    int kb = (lane < 16) ? 2 * p : 16 + 2 * p;
                    int K = ks * 32 + kb;
                    au[ks * 8 + p] = *(const unsigned int*)(hss + j * H_DIM + K);
                }
            }
        }

        // C chunk GEMM: 32 n-tiles per wave, K=64 -> 2 WMMAs per tile.
        // Ping-pong double buffer: two independent B register sets, no copies.
        v16bf b0a = *(const v16bf*)(fb + 0 * FRAG_ELEMS);
        v16bf b1a = *(const v16bf*)(fb + 1 * FRAG_ELEMS);
        v16bf b0b = *(const v16bf*)(fb + 2 * FRAG_ELEMS);
        v16bf b1b = *(const v16bf*)(fb + 3 * FRAG_ELEMS);
        #pragma unroll 1
        for (int t = 0; t < 32; t += 2) {
            // ---- even tile t: consume A-set, then refill A-set for t+2 ----
            v8f c = {};
            c = __builtin_amdgcn_wmma_f32_16x16x32_bf16(false, afr[0], false, b0a,
                                                        (short)0, c, false, false);
            c = __builtin_amdgcn_wmma_f32_16x16x32_bf16(false, afr[1], false, b1a,
                                                        (short)0, c, false, false);
            if (t + 2 < 32) {
                const __bf16* nf = fb + (t + 2) * 2 * FRAG_ELEMS;
                b0a = *(const v16bf*)(nf);
                b1a = *(const v16bf*)(nf + FRAG_ELEMS);
            }
            {
                int caddr = cbase32 + t * 16;
                #pragma unroll
                for (int k = 0; k < 4; ++k) {
                    v2bf pk; pk[0] = (__bf16)c[2 * k]; pk[1] = (__bf16)c[2 * k + 1];
                    unsigned int u; __builtin_memcpy(&u, &pk, 4);
                    Cs32[caddr + k * COLS] = u;
                }
            }
            // ---- odd tile t+1: consume B-set, then refill B-set for t+3 ----
            v8f d = {};
            d = __builtin_amdgcn_wmma_f32_16x16x32_bf16(false, afr[0], false, b0b,
                                                        (short)0, d, false, false);
            d = __builtin_amdgcn_wmma_f32_16x16x32_bf16(false, afr[1], false, b1b,
                                                        (short)0, d, false, false);
            if (t + 3 < 32) {
                const __bf16* nf = fb + (t + 3) * 2 * FRAG_ELEMS;
                b0b = *(const v16bf*)(nf);
                b1b = *(const v16bf*)(nf + FRAG_ELEMS);
            }
            {
                int caddr = cbase32 + (t + 1) * 16;
                #pragma unroll
                for (int k = 0; k < 4; ++k) {
                    v2bf pk; pk[0] = (__bf16)d[2 * k]; pk[1] = (__bf16)d[2 * k + 1];
                    unsigned int u; __builtin_memcpy(&u, &pk, 4);
                    Cs32[caddr + k * COLS] = u;
                }
            }
        }
        __syncthreads();

        // Gather-accumulate: out[i, ebase..+15] += C[jl, bin(i,jl)*64 + ebase ..]
        // Row pair p holds rows 2p (lo halves) and 2p+1 (hi halves).
        #pragma unroll 1
        for (int p = 0; p < 8; ++p) {
            int b0n = binsS[myi * 64 + jc * 16 + 2 * p];
            int b1n = binsS[myi * 64 + jc * 16 + 2 * p + 1];
            const unsigned int* rowp = Cs32 + p * COLS;
            if (b0n != 255) {
                const uint4* s4 = (const uint4*)(rowp + b0n * 64 + ebase);
                uint4 d0 = s4[0], d1 = s4[1], d2 = s4[2], d3 = s4[3];
                acc4_lo(acc + 0, d0);  acc4_lo(acc + 4,  d1);
                acc4_lo(acc + 8, d2);  acc4_lo(acc + 12, d3);
            }
            if (b1n != 255) {
                const uint4* s4 = (const uint4*)(rowp + b1n * 64 + ebase);
                uint4 d0 = s4[0], d1 = s4[1], d2 = s4[2], d3 = s4[3];
                acc4_hi(acc + 0, d0);  acc4_hi(acc + 4,  d1);
                acc4_hi(acc + 8, d2);  acc4_hi(acc + 12, d3);
            }
        }
        __syncthreads();   // before next chunk overwrites Cs32
    }

    // ---- bias + ReLU + store ----
    int ped = scene * SCENE_LEN + myi;
    #pragma unroll
    for (int e = 0; e < 16; ++e) {
        float o = acc[e] + bias[ebase + e];
        out[ped * EMB + ebase + e] = o > 0.0f ? o : 0.0f;
    }
}

extern "C" void kernel_launch(void* const* d_in, const int* in_sizes, int n_in,
                              void* d_out, int out_size, void* d_ws, size_t ws_size,
                              hipStream_t stream) {
    const float* hs   = (const float*)d_in[0];
    // d_in[1] = seq_start_end (int64) — scenes are fixed-size, unused
    const float* pos  = (const float*)d_in[2];
    const float* W    = (const float*)d_in[3];
    const float* bias = (const float*)d_in[4];
    float* out        = (float*)d_out;
    __bf16* wfrag     = (__bf16*)d_ws;   // needs NFRAGS*FRAG_ELEMS*2 = 512 KB

    // Repack W into bf16 WMMA B-fragments (deterministic, re-done every call).
    prep_w_frags<<<(NFRAGS * FRAG_ELEMS + 255) / 256, 256, 0, stream>>>(W, wfrag);

    // One workgroup (8 wave32) per scene.
    social_pool_wmma<<<N_SCENES, 256, 0, stream>>>(hs, pos, wfrag, bias, out);
}
